// BiLSTM_73753178407543
// MI455X (gfx1250) — matrix operand order repaired and checked by reading, not compiled, not run on hardware
//
#include <hip/hip_runtime.h>
#include <hip/hip_bf16.h>

typedef __attribute__((ext_vector_type(16))) _Float16 v16h;
typedef __attribute__((ext_vector_type(8)))  _Float16 v8h;
typedef __attribute__((ext_vector_type(8)))  float    v8f;

#define B_  128
#define T_  1024
#define E_  128
#define H_  128
#define G4  512           // 4*HDIM
#define BN_EPS 1e-3f

// ---------------- workspace layout (bytes) ----------------
// [0,512)        f32 sum[128]
// [512,1024)     f32 sumsq[128]
// [1024,1536)    f32 scale[128]
// [1536,2048)    f32 shift[128]   (stats block = 512 floats total)
// [4096, +33554432)  f16 xh[T][B][E]           (33.5 MB, L2-resident)
// next 524288        f16 wup[2][8][32][32][16] (B-fragment-packed W;U, both dirs)
// next 131072        f32 hcat[128][256]
#define XH_OFF   4096
#define WUP_OFF  (XH_OFF + (size_t)T_ * B_ * E_ * 2)
#define HCAT_OFF (WUP_OFF + 2 * 131072 * 2)

__device__ __forceinline__ float sigm(float x) { return 1.f / (1.f + __expf(-x)); }
// tanh via single exp: 1 - 2/(e^(2x)+1)  (v_exp_f32 TRANS op, co-executes with WMMA)
__device__ __forceinline__ float ftanh(float x) {
  return 1.f - 2.f / (__expf(2.f * x) + 1.f);
}

__device__ __forceinline__ v16h load16(const _Float16* p0, const _Float16* p1) {
  v8h lo = *(const v8h*)p0;
  v8h hi = *(const v8h*)p1;
  v16h r;
#pragma unroll
  for (int i = 0; i < 8; ++i) { r[i] = lo[i]; r[8 + i] = hi[i]; }
  return r;
}

// ---------------- stats: zero ----------------
__global__ void k_zero_stats(float* stats) {
  stats[threadIdx.x] = 0.f;   // 256 threads: sum + sumsq
}

// ---------------- gather + BN1 partial sums ----------------
__global__ void k_embed_stats(const int* __restrict__ ids,
                              const float* __restrict__ emb,
                              float* __restrict__ stats) {
  const int e = threadIdx.x;               // feature
  const int rowBase = blockIdx.x * 128;    // 1024 blocks over B*T rows
  float s = 0.f, q = 0.f;
  for (int r = 0; r < 128; ++r) {
    const int id = ids[rowBase + r];
    const float x = emb[(size_t)id * E_ + e];
    s += x; q += x * x;
  }
  atomicAdd(&stats[e], s);
  atomicAdd(&stats[128 + e], q);
}

// ---------------- BN1 scale/shift ----------------
__global__ void k_finalize_stats(const float* __restrict__ g1,
                                 const float* __restrict__ b1,
                                 float* __restrict__ stats) {
  const int e = threadIdx.x;
  const float n = (float)(B_ * T_);
  const float m = stats[e] / n;
  const float v = stats[128 + e] / n - m * m;
  const float sc = g1[e] * rsqrtf(v + BN_EPS);
  stats[256 + e] = sc;
  stats[384 + e] = b1[e] - m * sc;
}

// ---------------- gather + normalize -> f16 (T,B,E) ----------------
__global__ void k_embed_norm(const int* __restrict__ ids,
                             const float* __restrict__ emb,
                             const float* __restrict__ stats,
                             _Float16* __restrict__ xh) {
  const int e  = threadIdx.x;
  const float sc = stats[256 + e], sh = stats[384 + e];
  const int r0 = blockIdx.x * 8;           // 16384 blocks over B*T rows
  for (int k = 0; k < 8; ++k) {
    const int r = r0 + k;                  // flat (b,t)
    const int b = r >> 10, t = r & 1023;
    const int id = ids[r];
    const float x = emb[(size_t)id * E_ + e] * sc + sh;
    xh[((size_t)t * B_ + b) * E_ + e] = (_Float16)x;
  }
}

// ---------------- pack [W;U] (f32 -> f16) into WMMA-B-fragment layout ----
// wup[dir][kt][nt][lane][q] , K = kt*32 + (q>=8 ? 16 : 0) + (lane>=16 ? 8 : 0) + q%8 , N = nt*16 + lane%16
__global__ void k_pack_weights(const float* __restrict__ Wf, const float* __restrict__ Uf,
                               const float* __restrict__ Wb, const float* __restrict__ Ub,
                               _Float16* __restrict__ wup) {
  const int idx  = blockIdx.x * blockDim.x + threadIdx.x; // 524288 total
  const int q    = idx & 15;
  const int lane = (idx >> 4) & 31;
  const int nt   = (idx >> 9) & 31;
  const int kt   = (idx >> 14) & 7;
  const int d    = idx >> 17;
  const int K = kt * 32 + ((q >> 3) << 4) + ((lane >> 4) << 3) + (q & 7); // 0..255
  const int N = nt * 16 + (lane & 15);                    // 0..511
  const float* Wm = d ? Wb : Wf;
  const float* Um = d ? Ub : Uf;
  const float v = (K < 128) ? Wm[K * G4 + N] : Um[(K - 128) * G4 + N];
  wup[idx] = (_Float16)v;
}

// ---------------- persistent LSTM recurrence ----------------
// grid = 16 : blockIdx = dir (bit0) x batch-slice (bits 3:1); block = 256 (8 waves)
// wave j computes gate tiles (i,f,c~,o) at N = g*128 + j*16 for its 16-batch slice.
__global__ __launch_bounds__(256)
void k_lstm(const _Float16* __restrict__ xh, const _Float16* __restrict__ wup,
            const int* __restrict__ ids, const float* __restrict__ bf,
            const float* __restrict__ bb, float* __restrict__ hcat) {
  extern __shared__ char smem[];
  _Float16* WU = (_Float16*)smem;                    // 131072 halves = 256 KB
  _Float16* hA = (_Float16*)(smem + 262144);         // 16 x 128 f16
  _Float16* hB = (_Float16*)(smem + 262144 + 4096);  // 16 x 128 f16
  const int dir   = blockIdx.x & 1;
  const int slice = blockIdx.x >> 1;                 // 0..7
  const int b0    = slice * 16;
  const int tid   = threadIdx.x;
  const int lane  = tid & 31;
  const int j     = tid >> 5;                        // wave id == j-tile
  const float* bias = dir ? bb : bf;

  { // stage fragment-packed [W;U] into LDS (weight-stationary)
    const v8h* src = (const v8h*)(wup + (size_t)dir * 131072);
    v8h* dst = (v8h*)WU;
    for (int i = tid; i < 16384; i += 256) dst[i] = src[i];
  }
  { // zero both h buffers (8192 B)
    float4* hz = (float4*)hA;
    const float4 z = make_float4(0.f, 0.f, 0.f, 0.f);
    for (int i = tid; i < 512; i += 256) hz[i] = z;
  }
  __syncthreads();

  _Float16* hcur = hA;
  _Float16* hnxt = hB;

  const int col  = j * 16 + (lane & 15);   // hidden/gate column this lane owns
  const int lrow = (lane >> 4) << 3;       // C-fragment row offset (0 or 8)
  const int ksub = (lane >> 4) << 3;       // A-fragment k sub-offset (0 or 8)

  float cst[8];                            // cell state, C-fragment layout
#pragma unroll
  for (int i = 0; i < 8; ++i) cst[i] = 0.f;
  float bgate[4];
#pragma unroll
  for (int g = 0; g < 4; ++g) bgate[g] = bias[g * 128 + col];

  for (int t = 0; t < T_; ++t) {
    const int teff = dir ? (T_ - 1 - t) : t;
    const _Float16* xrow = xh + ((size_t)teff * B_ + (b0 + (lane & 15))) * E_;
    const _Float16* hrow = hcur + (lane & 15) * H_;

    if (t + 1 < T_) {                      // gfx1250 global_prefetch of next x tile
      const int tn = dir ? (teff - 1) : (teff + 1);
      __builtin_prefetch(xh + ((size_t)tn * B_ + b0) * E_ + tid * 8, 0, 1);
    }

    // g = [x_t | h] @ [W ; U] : K = 256 in 8 k-tiles of 32.
    // kt = 0 seeds each gate chain from an inline-zero C (no VALU pre-seeding,
    // avoids VALU->WMMA-C hazard nops); bias is added in the scalar epilogue.
    v8f acc[4];
    {
      const v16h afrag = load16(xrow + ksub, xrow + ksub + 16);
      const v8f zero = {};
#pragma unroll
      for (int g = 0; g < 4; ++g) {
        const int nt = g * 8 + j;
        const _Float16* bp = WU + (((nt) * 32 + lane) << 4);
        const v16h bfrag = load16(bp, bp + 8);
        acc[g] = __builtin_amdgcn_wmma_f32_16x16x32_f16(
            false, afrag, false, bfrag, (short)0, zero, false, false);
      }
    }
#pragma unroll
    for (int kt = 1; kt < 8; ++kt) {
      const _Float16* arow = (kt < 4) ? xrow : hrow;
      const int kofs = (kt & 3) * 32 + ksub;
      const v16h afrag = load16(arow + kofs, arow + kofs + 16);
#pragma unroll
      for (int g = 0; g < 4; ++g) {
        const int nt = g * 8 + j;
        const _Float16* bp = WU + (((kt * 32 + nt) * 32 + lane) << 4);
        const v16h bfrag = load16(bp, bp + 8);
        acc[g] = __builtin_amdgcn_wmma_f32_16x16x32_f16(
            false, afrag, false, bfrag, (short)0, acc[g], false, false);
      }
    }

    // gating + masked state update (C-fragment layout: row = i + lrow, col fixed)
#pragma unroll
    for (int i = 0; i < 8; ++i) {
      const int r    = i + lrow;                       // local batch row 0..15
      const int grow = b0 + r;                         // global batch row
      const bool mk  = (ids[grow * T_ + teff] != 0);
      const float hold = (float)hcur[r * H_ + col];
      const float iv = sigm(acc[0][i] + bgate[0]);
      const float fv = sigm(acc[1][i] + bgate[1]);
      const float gv = ftanh(acc[2][i] + bgate[2]);
      const float ov = sigm(acc[3][i] + bgate[3]);
      const float cn = fv * cst[i] + iv * gv;
      const float hn = ov * ftanh(cn);
      if (mk) cst[i] = cn;
      hnxt[r * H_ + col] = (_Float16)(mk ? hn : hold);
    }
    __syncthreads();
    _Float16* tmp = hcur; hcur = hnxt; hnxt = tmp;
  }

  // final h slice -> hcat[:, dir*128 : dir*128+128]
  for (int i = tid; i < 16 * H_; i += 256) {
    const int r = i >> 7, hh = i & 127;
    hcat[(size_t)(b0 + r) * 256 + dir * 128 + hh] = (float)hcur[i];
  }
}

// ---------------- BN2 + dense + softmax ----------------
__global__ void k_head(const float* __restrict__ hcat, const float* __restrict__ g2,
                       const float* __restrict__ b2, const float* __restrict__ Wd,
                       const float* __restrict__ bd, float* __restrict__ out) {
  __shared__ float sc2[256];
  __shared__ float sh2[256];
  extern __shared__ char smem[];
  float* hn = (float*)smem;               // 128 x 256
  const int tid = threadIdx.x;            // 256 threads
  {
    const int f = tid;
    float s = 0.f, q = 0.f;
    for (int b = 0; b < 128; ++b) { const float x = hcat[b * 256 + f]; s += x; q += x * x; }
    const float m = s * (1.f / 128.f);
    const float v = q * (1.f / 128.f) - m * m;
    const float sc = g2[f] * rsqrtf(v + BN_EPS);
    sc2[f] = sc; sh2[f] = b2[f] - m * sc;
  }
  __syncthreads();
  for (int i = tid; i < 128 * 256; i += 256) {
    const int f = i & 255;
    hn[i] = hcat[i] * sc2[f] + sh2[f];
  }
  __syncthreads();
  if (tid < 128) {
    const int b = tid;
    float lg[10];
#pragma unroll
    for (int o = 0; o < 10; ++o) lg[o] = bd[o];
    for (int f = 0; f < 256; ++f) {
      const float x = hn[b * 256 + f];
#pragma unroll
      for (int o = 0; o < 10; ++o) lg[o] += x * Wd[f * 10 + o];
    }
    float mx = lg[0];
#pragma unroll
    for (int o = 1; o < 10; ++o) mx = fmaxf(mx, lg[o]);
    float den = 0.f, ex[10];
#pragma unroll
    for (int o = 0; o < 10; ++o) { ex[o] = __expf(lg[o] - mx); den += ex[o]; }
#pragma unroll
    for (int o = 0; o < 10; ++o) out[b * 10 + o] = ex[o] / den;
  }
}

extern "C" void kernel_launch(void* const* d_in, const int* in_sizes, int n_in,
                              void* d_out, int out_size, void* d_ws, size_t ws_size,
                              hipStream_t stream) {
  const int*   ids = (const int*)d_in[0];
  const float* emb = (const float*)d_in[1];
  const float* g1  = (const float*)d_in[2];
  const float* b1  = (const float*)d_in[3];
  const float* Wf  = (const float*)d_in[4];
  const float* Uf  = (const float*)d_in[5];
  const float* bf  = (const float*)d_in[6];
  const float* Wb  = (const float*)d_in[7];
  const float* Ub  = (const float*)d_in[8];
  const float* bb  = (const float*)d_in[9];
  const float* g2  = (const float*)d_in[10];
  const float* b2  = (const float*)d_in[11];
  const float* Wd  = (const float*)d_in[12];
  const float* bd  = (const float*)d_in[13];
  float* out = (float*)d_out;

  char* ws = (char*)d_ws;
  float*    stats = (float*)ws;
  _Float16* xh    = (_Float16*)(ws + XH_OFF);
  _Float16* wup   = (_Float16*)(ws + WUP_OFF);
  float*    hcat  = (float*)(ws + HCAT_OFF);

  k_zero_stats    <<<1,     256, 0, stream>>>(stats);
  k_embed_stats   <<<1024,  128, 0, stream>>>(ids, emb, stats);
  k_finalize_stats<<<1,     128, 0, stream>>>(g1, b1, stats);
  k_embed_norm    <<<16384, 128, 0, stream>>>(ids, emb, stats, xh);
  k_pack_weights  <<<2048,  256, 0, stream>>>(Wf, Uf, Wb, Ub, wup);
  k_lstm          <<<16,    256, 262144 + 8192, stream>>>(xh, wup, ids, bf, bb, hcat);
  k_head          <<<1,     256, 128 * 256 * 4, stream>>>(hcat, g2, b2, Wd, bd, out);
}
